// DistributedGNN_53455162966375
// MI455X (gfx1250) — matrix operand order repaired and checked by reading, not compiled, not run on hardware
//
#include <hip/hip_runtime.h>
#include <math.h>

typedef __bf16 bf16;
typedef __attribute__((ext_vector_type(16))) __bf16 v16bf;
typedef __attribute__((ext_vector_type(8)))  __bf16 v8bf;
typedef __attribute__((ext_vector_type(8)))  float  v8f;

typedef __attribute__((address_space(3))) bf16 lbf16;
typedef __attribute__((address_space(3))) const v8bf* v8p_lds;

#define LN_EPS 1e-5f
#define LDS_PAD 8   // bf16 elements of row padding -> bank-conflict-free B reads

// ---------- small helper kernels ----------

__global__ void cvt_f2b_kernel(const float* __restrict__ s, bf16* __restrict__ d, int n) {
  int i = blockIdx.x * blockDim.x + threadIdx.x;
  if (i < n) d[i] = (bf16)s[i];
}

__global__ void zero_f_kernel(float* __restrict__ p, int n) {
  int i = blockIdx.x * blockDim.x + threadIdx.x;
  if (i < n) p[i] = 0.f;
}

// first layer of encoders: din = 3 or 4, dout = 128, ELU, bf16 out
__global__ void enc_first_kernel(const float* __restrict__ X, int din,
                                 const float* __restrict__ W, const float* __restrict__ b,
                                 int rows, bf16* __restrict__ out) {
  int gid = blockIdx.x * blockDim.x + threadIdx.x;
  if (gid >= rows * 128) return;
  int r = gid >> 7, c = gid & 127;
  float s = b[c];
  for (int k = 0; k < din; ++k) s += X[r * din + k] * W[c * din + k];
  s = s > 0.f ? s : expf(s) - 1.f;
  out[gid] = (bf16)s;
}

// last decoder layer: din = 128, dout = 3, fp32 out
__global__ void dec_last_kernel(const bf16* __restrict__ X, const float* __restrict__ W,
                                const float* __restrict__ b, int rows, float* __restrict__ out) {
  int gid = blockIdx.x * blockDim.x + threadIdx.x;
  if (gid >= rows * 3) return;
  int r = gid / 3, c = gid - r * 3;
  float s = b[c];
  const bf16* xr = X + (size_t)r * 128;
  const float* wr = W + c * 128;
  for (int k = 0; k < 128; ++k) s += (float)xr[k] * wr[k];
  out[gid] = s;
}

// ---------- generic fused WMMA MLP layer ----------
// Y[rows x 128] = act( concat-A[rows x (d0+d1+d2)] @ W^T + b ), W is (128 x din) bf16 row-major.
// W is staged into LDS once per workgroup via async global->LDS copies.
// Each wave computes a 32-row x 128-col tile: two 16-row A tiles share every
// B fragment (halves LDS read bytes per WMMA, doubles WMMA ILP).
// mode 0: ELU -> obf
// mode 1: LayerNorm(g,be) [+resid] [*ew] -> of32/obf, optional atomicAdd scatter by sidx.
__global__ __launch_bounds__(256) void mlp_wmma_kernel(
    const bf16* __restrict__ A0, const int* __restrict__ I0, int d0,
    const bf16* __restrict__ A1, const int* __restrict__ I1, int d1,
    const bf16* __restrict__ A2, const int* __restrict__ I2, int d2,
    const bf16* __restrict__ W, const float* __restrict__ bias,
    int rows, int mode,
    const float* __restrict__ g, const float* __restrict__ be,
    const float* __restrict__ resid, const float* __restrict__ ew,
    float* __restrict__ adst, const int* __restrict__ sidx,
    float* __restrict__ of32, bf16* __restrict__ obf) {
  extern __shared__ char smem[];
  lbf16* lw = (lbf16*)smem;                 // 128 x (din + LDS_PAD) bf16

  const int lane = threadIdx.x & 31;
  const int wave = threadIdx.x >> 5;
  const int tile = blockIdx.x * 8 + wave;   // 32-row tile id
  const int l15  = lane & 15;
  const bool hi  = lane >= 16;
  const int din  = d0 + d1 + d2;
  const int lstride = din + LDS_PAD;

  // ---- stage W into LDS with async global->LDS copies (ASYNCcnt) ----
  {
    const int din8 = din >> 3;              // 16-byte chunks per row
    for (int row = wave; row < 128; row += 8) {
      const bf16* grow = W + (size_t)row * din;
      lbf16* lrow = lw + (size_t)row * lstride;
      for (int c = lane; c < din8; c += 32) {
        unsigned laddr = (unsigned)(size_t)(lrow + c * 8);
        const bf16* gp = grow + c * 8;
        asm volatile("global_load_async_to_lds_b128 %0, %1, off"
                     :: "v"(laddr), "v"(gp)
                     : "memory");
      }
    }
    asm volatile("s_wait_asynccnt 0" ::: "memory");
    __syncthreads();
  }

  // A rows: lane owns rows tile*32 + (lane&15) and +16; halves of the wave split K.
  int rA0 = tile * 32 + l15;
  int rA1 = rA0 + 16;
  if (rA0 > rows - 1) rA0 = rows - 1;
  if (rA1 > rows - 1) rA1 = rows - 1;

  const bf16* b0a = A0 + (size_t)(I0 ? I0[rA0] : rA0) * (size_t)d0;
  const bf16* b0b = A0 + (size_t)(I0 ? I0[rA1] : rA1) * (size_t)d0;
  const bf16* b1a = A1 ? (A1 + (size_t)(I1 ? I1[rA0] : rA0) * (size_t)d1) : (const bf16*)0;
  const bf16* b1b = A1 ? (A1 + (size_t)(I1 ? I1[rA1] : rA1) * (size_t)d1) : (const bf16*)0;
  const bf16* b2a = A2 ? (A2 + (size_t)(I2 ? I2[rA0] : rA0) * (size_t)d2) : (const bf16*)0;
  const bf16* b2b = A2 ? (A2 + (size_t)(I2 ? I2[rA1] : rA1) * (size_t)d2) : (const bf16*)0;

  const int kbA = hi ? 8 : 0;    // A: lanes 16-31 hold K 8-15 / 24-31
  const int kbB = hi ? 8 : 0;    // B: lanes 16-31 hold K 8-15 / 24-31
  const lbf16* lwB = lw + (size_t)l15 * lstride + kbB;   // B col (N=l15) base in LDS

  v8f acc[2][8];
#pragma unroll
  for (int t = 0; t < 2; ++t)
#pragma unroll
    for (int n = 0; n < 8; ++n)
#pragma unroll
      for (int i = 0; i < 8; ++i) acc[t][n][i] = 0.f;

  // one k-step (K = 32): two A fragments from global, each B fragment feeds 2 WMMAs
  auto do_kstep = [&](const bf16* pA0, const bf16* pA1, int k0) {
    v8bf a00 = *(const v8bf*)(pA0 + kbA);
    v8bf a01 = *(const v8bf*)(pA0 + kbA + 16);
    v8bf a10 = *(const v8bf*)(pA1 + kbA);
    v8bf a11 = *(const v8bf*)(pA1 + kbA + 16);
    v16bf Am0, Am1;
#pragma unroll
    for (int i = 0; i < 8; ++i) {
      Am0[i] = a00[i]; Am0[i + 8] = a01[i];
      Am1[i] = a10[i]; Am1[i + 8] = a11[i];
    }
#pragma unroll
    for (int n = 0; n < 8; ++n) {
      const lbf16* pb = lwB + n * 16 * lstride + k0;
      v8bf bv0 = *(v8p_lds)(pb);
      v8bf bv1 = *(v8p_lds)(pb + 16);
      v16bf Bm;
#pragma unroll
      for (int i = 0; i < 8; ++i) { Bm[i] = bv0[i]; Bm[i + 8] = bv1[i]; }
      acc[0][n] = __builtin_amdgcn_wmma_f32_16x16x32_bf16(
          false, Am0, false, Bm, (short)0, acc[0][n], false, false);
      acc[1][n] = __builtin_amdgcn_wmma_f32_16x16x32_bf16(
          false, Am1, false, Bm, (short)0, acc[1][n], false, false);
    }
  };

  // branch-free segment walk (each segment width is a multiple of 32)
  int kacc = 0;
  for (int kk = 0; kk < d0; kk += 32) do_kstep(b0a + kk, b0b + kk, kacc + kk);
  kacc += d0;
  if (d1) {
    for (int kk = 0; kk < d1; kk += 32) do_kstep(b1a + kk, b1b + kk, kacc + kk);
    kacc += d1;
  }
  if (d2) {
    for (int kk = 0; kk < d2; kk += 32) do_kstep(b2a + kk, b2b + kk, kacc + kk);
  }

  float bcol[8];
#pragma unroll
  for (int n = 0; n < 8; ++n) bcol[n] = bias[n * 16 + l15];

  if (mode == 0) {  // ELU -> bf16
#pragma unroll
    for (int t = 0; t < 2; ++t) {
#pragma unroll
      for (int r = 0; r < 8; ++r) {
        int row = tile * 32 + t * 16 + r + (hi ? 8 : 0);
        if (row < rows) {
#pragma unroll
          for (int n = 0; n < 8; ++n) {
            float y = acc[t][n][r] + bcol[n];
            y = y > 0.f ? y : expf(y) - 1.f;
            obf[(size_t)row * 128 + n * 16 + l15] = (bf16)y;
          }
        }
      }
    }
  } else {          // LayerNorm epilogue
    float gcol[8], becol[8];
#pragma unroll
    for (int n = 0; n < 8; ++n) { gcol[n] = g[n * 16 + l15]; becol[n] = be[n * 16 + l15]; }
#pragma unroll
    for (int t = 0; t < 2; ++t) {
#pragma unroll
      for (int r = 0; r < 8; ++r) {
        float vals[8], s = 0.f, q = 0.f;
#pragma unroll
        for (int n = 0; n < 8; ++n) {
          float v = acc[t][n][r] + bcol[n];
          vals[n] = v; s += v; q += v * v;
        }
        // row stats: butterfly across the 16 lanes of this half (wave32 halves)
#pragma unroll
        for (int off = 1; off < 16; off <<= 1) {
          s += __shfl_xor(s, off, 32);
          q += __shfl_xor(q, off, 32);
        }
        float mean = s * (1.f / 128.f);
        float var  = q * (1.f / 128.f) - mean * mean;
        float rstd = rsqrtf(var + LN_EPS);
        int row = tile * 32 + t * 16 + r + (hi ? 8 : 0);
        if (row < rows) {
          float ewv = ew ? ew[row] : 1.f;
          int srow = sidx ? sidx[row] : 0;
#pragma unroll
          for (int n = 0; n < 8; ++n) {
            int col = n * 16 + l15;
            float y = (vals[n] - mean) * rstd * gcol[n] + becol[n];
            if (resid) y += resid[(size_t)row * 128 + col];
            y *= ewv;
            if (of32) of32[(size_t)row * 128 + col] = y;
            if (obf)  obf[(size_t)row * 128 + col]  = (bf16)y;
            if (adst) atomicAdd(adst + (size_t)srow * 128 + col, y);
          }
        }
      }
    }
  }
}

// ---------- host orchestration ----------

extern "C" void kernel_launch(void* const* d_in, const int* in_sizes, int n_in,
                              void* d_out, int out_size, void* d_ws, size_t ws_size,
                              hipStream_t stream) {
  (void)n_in; (void)out_size; (void)ws_size;
  const float* X0 = (const float*)d_in[0];
  const int*   EI = (const int*)d_in[1];
  const float* EA = (const float*)d_in[2];
  const float* EW = (const float*)d_in[3];

  const int N = in_sizes[0] / 3;
  const int E = in_sizes[3];
  const int* src = EI;
  const int* dst = EI + E;

  auto F = [&](int i) { return (const float*)d_in[i]; };
  // JAX pytree (sorted dict keys) leaf order:
  // 4..9   dec    W0 b0 W1 b1 W2 b2
  // 10..17 edge_enc W0 b0 W1 b1 W2 b2 g be
  // 18..25 mp0.edge  W0 b0 W1 b1 W2 b2 g be
  // 26..33 mp0.node  W0 b0 W1 b1 W2 b2 g be
  // 34..41 mp1.edge ; 42..49 mp1.node
  // 50..57 node_enc W0 b0 W1 b1 W2 b2 g be

  // allow >64KB dynamic LDS for the 128x384 layer (WGP has 320KB)
  hipFuncSetAttribute(reinterpret_cast<const void*>(mlp_wmma_kernel),
                      hipFuncAttributeMaxDynamicSharedMemorySize,
                      128 * (384 + LDS_PAD) * 2);

  char* ws = (char*)d_ws;
  size_t off = 0;
  auto alloc = [&](size_t bytes) -> char* {
    char* p = ws + off;
    off += (bytes + 255) & ~(size_t)255;
    return p;
  };
  auto cvtW = [&](int idx, int elems) -> bf16* {
    bf16* p = (bf16*)alloc((size_t)elems * sizeof(bf16));
    cvt_f2b_kernel<<<(elems + 255) / 256, 256, 0, stream>>>(F(idx), p, elems);
    return p;
  };

  // bf16 weight copies (deterministic each call)
  bf16* decW0b = cvtW(4, 128 * 128);
  bf16* decW1b = cvtW(6, 128 * 128);
  bf16* eeW1b  = cvtW(12, 128 * 128);
  bf16* eeW2b  = cvtW(14, 128 * 128);
  bf16* neW1b  = cvtW(52, 128 * 128);
  bf16* neW2b  = cvtW(54, 128 * 128);
  bf16 *mpeW0b[2], *mpeW1b[2], *mpeW2b[2], *mpnW0b[2], *mpnW1b[2], *mpnW2b[2];
  for (int r = 0; r < 2; ++r) {
    int eb = 18 + r * 16, nb = 26 + r * 16;
    mpeW0b[r] = cvtW(eb + 0, 128 * 384);
    mpeW1b[r] = cvtW(eb + 2, 128 * 128);
    mpeW2b[r] = cvtW(eb + 4, 128 * 128);
    mpnW0b[r] = cvtW(nb + 0, 128 * 256);
    mpnW1b[r] = cvtW(nb + 2, 128 * 128);
    mpnW2b[r] = cvtW(nb + 4, 128 * 128);
  }

  // activation buffers
  float* x_f  = (float*)alloc((size_t)N * 128 * 4);
  bf16*  x_b  = (bf16*) alloc((size_t)N * 128 * 2);
  float* e_f  = (float*)alloc((size_t)E * 128 * 4);
  bf16*  e_b  = (bf16*) alloc((size_t)E * 128 * 2);
  bf16*  h1   = (bf16*) alloc((size_t)E * 128 * 2);
  bf16*  h2   = (bf16*) alloc((size_t)E * 128 * 2);
  float* agg  = (float*)alloc((size_t)N * 128 * 4);
  bf16*  aggb = (bf16*) alloc((size_t)N * 128 * 2);

  auto layer = [&](const bf16* A0, const int* I0, int d0,
                   const bf16* A1, const int* I1, int d1,
                   const bf16* A2, const int* I2, int d2,
                   const bf16* Wb, const float* bias, int rowsL, int mode,
                   const float* gg, const float* bb, const float* resid, const float* eww,
                   float* adst, const int* sidxL, float* of32, bf16* obf) {
    int tiles = (rowsL + 31) / 32;           // 32 rows per wave
    int blocks = (tiles + 7) / 8;            // 8 waves per block
    int din = d0 + d1 + d2;
    size_t sh = (size_t)128 * (din + LDS_PAD) * 2;
    mlp_wmma_kernel<<<blocks, 256, sh, stream>>>(
        A0, I0, d0, A1, I1, d1, A2, I2, d2, Wb, bias, rowsL, mode,
        gg, bb, resid, eww, adst, sidxL, of32, obf);
  };

  // ---- node encoder: 3 -> 128 -> 128 -> 128 + LN ----
  enc_first_kernel<<<(N * 128 + 255) / 256, 256, 0, stream>>>(X0, 3, F(50), F(51), N, h1);
  layer(h1, nullptr, 128, nullptr, nullptr, 0, nullptr, nullptr, 0,
        neW1b, F(53), N, 0, nullptr, nullptr, nullptr, nullptr, nullptr, nullptr, nullptr, h2);
  layer(h2, nullptr, 128, nullptr, nullptr, 0, nullptr, nullptr, 0,
        neW2b, F(55), N, 1, F(56), F(57), nullptr, nullptr, nullptr, nullptr, x_f, x_b);

  // ---- edge encoder: 4 -> 128 -> 128 -> 128 + LN ----
  enc_first_kernel<<<(E * 128 + 255) / 256, 256, 0, stream>>>(EA, 4, F(10), F(11), E, h1);
  layer(h1, nullptr, 128, nullptr, nullptr, 0, nullptr, nullptr, 0,
        eeW1b, F(13), E, 0, nullptr, nullptr, nullptr, nullptr, nullptr, nullptr, nullptr, h2);
  layer(h2, nullptr, 128, nullptr, nullptr, 0, nullptr, nullptr, 0,
        eeW2b, F(15), E, 1, F(16), F(17), nullptr, nullptr, nullptr, nullptr, e_f, e_b);

  // ---- 2 message-passing rounds ----
  for (int r = 0; r < 2; ++r) {
    int eb = 18 + r * 16, nb = 26 + r * 16;
    zero_f_kernel<<<(N * 128 + 255) / 256, 256, 0, stream>>>(agg, N * 128);
    // edge MLP on concat([x[src], x[dst], e]) : 384 -> 128 -> 128 -> 128
    layer(x_b, src, 128, x_b, dst, 128, e_b, nullptr, 128,
          mpeW0b[r], F(eb + 1), E, 0, nullptr, nullptr, nullptr, nullptr,
          nullptr, nullptr, nullptr, h1);
    layer(h1, nullptr, 128, nullptr, nullptr, 0, nullptr, nullptr, 0,
          mpeW1b[r], F(eb + 3), E, 0, nullptr, nullptr, nullptr, nullptr,
          nullptr, nullptr, nullptr, h2);
    // LN + residual(e) + *ew, fused segment-sum scatter into agg
    layer(h2, nullptr, 128, nullptr, nullptr, 0, nullptr, nullptr, 0,
          mpeW2b[r], F(eb + 5), E, 1, F(eb + 6), F(eb + 7), e_f, EW,
          agg, dst, e_f, e_b);
    cvt_f2b_kernel<<<(N * 128 + 255) / 256, 256, 0, stream>>>(agg, aggb, N * 128);
    // node MLP on concat([x, agg]) : 256 -> 128 -> 128 -> 128, LN + residual(x)
    layer(x_b, nullptr, 128, aggb, nullptr, 128, nullptr, nullptr, 0,
          mpnW0b[r], F(nb + 1), N, 0, nullptr, nullptr, nullptr, nullptr,
          nullptr, nullptr, nullptr, h1);
    layer(h1, nullptr, 128, nullptr, nullptr, 0, nullptr, nullptr, 0,
          mpnW1b[r], F(nb + 3), N, 0, nullptr, nullptr, nullptr, nullptr,
          nullptr, nullptr, nullptr, h2);
    layer(h2, nullptr, 128, nullptr, nullptr, 0, nullptr, nullptr, 0,
          mpnW2b[r], F(nb + 5), N, 1, F(nb + 6), F(nb + 7), x_f, nullptr,
          nullptr, nullptr, x_f, x_b);
  }

  // ---- decoder: 128 -> 128 -> 128 -> 3 ----
  layer(x_b, nullptr, 128, nullptr, nullptr, 0, nullptr, nullptr, 0,
        decW0b, F(5), N, 0, nullptr, nullptr, nullptr, nullptr, nullptr, nullptr, nullptr, h1);
  layer(h1, nullptr, 128, nullptr, nullptr, 0, nullptr, nullptr, 0,
        decW1b, F(7), N, 0, nullptr, nullptr, nullptr, nullptr, nullptr, nullptr, nullptr, h2);
  dec_last_kernel<<<(N * 3 + 255) / 256, 256, 0, stream>>>(h2, F(8), F(9), N, (float*)d_out);
}
